// LinearTransformerEncoderLayer_76020921139202
// MI455X (gfx1250) — matrix operand order repaired and checked by reading, not compiled, not run on hardware
//
#include <hip/hip_runtime.h>
#include <hip/hip_bf16.h>
#include <math.h>

// ---------------------------------------------------------------------------
// Types / constants
// ---------------------------------------------------------------------------
typedef _Float16 f16x8  __attribute__((ext_vector_type(8)));
typedef _Float16 f16x16 __attribute__((ext_vector_type(16)));
typedef float    f32x8  __attribute__((ext_vector_type(8)));

namespace {
constexpr int Bn  = 4;
constexpr int Tn  = 4096;
constexpr int Dn  = 1024;
constexpr int Hn  = 16;
constexpr int HDn = 64;
constexpr int FFn = 4096;
constexpr int MTOK = Bn * Tn;          // 16384 token rows
constexpr float EPS_ATTN = 1e-6f;
constexpr float EPS_LN   = 1e-5f;
constexpr int LDS_PITCH = 40;          // halves per row: 80B rows, bank-friendly
}

#define SHUF16(lo, hi) __builtin_shufflevector((lo), (hi), 0,1,2,3,4,5,6,7,8,9,10,11,12,13,14,15)

// A-fragment (16x32 f16): lane m=lane&15, hi=lane>>4.
// elements 0..7 = K(hi*8 .. hi*8+7), elements 8..15 = K(hi*8+16 .. hi*8+23)
__device__ __forceinline__ f16x16 load_frag_a(const _Float16* p) {
  f16x8 lo = *(const f16x8*)p;
  f16x8 hh = *(const f16x8*)(p + 16);
  return SHUF16(lo, hh);
}
// B-fragment (32x16 f16): lane n=lane&15, hi=lane>>4 holds K(hi*8 .. hi*8+7), K(hi*8+16 .. hi*8+23),
// when B is stored transposed (row n over K).
__device__ __forceinline__ f16x16 load_frag_b(const _Float16* p) {
  f16x8 lo = *(const f16x8*)p;
  f16x8 hh = *(const f16x8*)(p + 16);
  return SHUF16(lo, hh);
}

// ---------------------------------------------------------------------------
// CDNA5 async global -> LDS (ASYNCcnt-tracked, bypasses VGPRs)
// ---------------------------------------------------------------------------
__device__ __forceinline__ unsigned lds_off_of(const void* p) {
  return (unsigned)(size_t)(__attribute__((address_space(3))) const void*)p;
}
__device__ __forceinline__ void async_ld128(unsigned lds_off, const _Float16* g) {
  asm volatile("global_load_async_to_lds_b128 %0, %1, off"
               :: "v"(lds_off), "v"((unsigned long long)(size_t)g)
               : "memory");
}
__device__ __forceinline__ void wait_async0() {
  asm volatile("s_wait_asynccnt 0" ::: "memory");
}

// ---------------------------------------------------------------------------
// Weight convert + transpose: Wt[n*K + k] = (f16) W[k*N + n]
// ---------------------------------------------------------------------------
__global__ __launch_bounds__(256) void transpose_f32_to_f16(
    const float* __restrict__ W, _Float16* __restrict__ Wt, int K, int N) {
  __shared__ float tile[32][33];
  const int n0 = blockIdx.x * 32, k0 = blockIdx.y * 32;
  const int tx = threadIdx.x, ty = threadIdx.y;   // block (32,8)
  for (int i = 0; i < 4; i++)
    tile[ty + i * 8][tx] = W[(size_t)(k0 + ty + i * 8) * N + n0 + tx];
  __syncthreads();
  for (int i = 0; i < 4; i++)
    Wt[(size_t)(n0 + ty + i * 8) * K + k0 + tx] = (_Float16)tile[tx][ty + i * 8];
}

__global__ __launch_bounds__(256) void f32_to_f16_kernel(
    const float* __restrict__ x, _Float16* __restrict__ y) {
  const size_t i = ((size_t)blockIdx.x * 256 + threadIdx.x) * 4;
  y[i + 0] = (_Float16)x[i + 0];
  y[i + 1] = (_Float16)x[i + 1];
  y[i + 2] = (_Float16)x[i + 2];
  y[i + 3] = (_Float16)x[i + 3];
}

// ---------------------------------------------------------------------------
// Generic WMMA GEMM: C[M,N] = act(A[M,K](f16) @ Bt[N,K]^T(f16) + bias) (+resid)
// Block tile 128x128, 8 waves (4 x 2), wave tile 32x64 (2x4 WMMA frags).
// Double-buffered LDS, tiles staged with global_load_async_to_lds_b128 so the
// next K-slice streams in while WMMAs run; one barrier per K-step.
// ACT: 0=none 1=relu 2=elu+1
// ---------------------------------------------------------------------------
template <int ACT, bool HALF_OUT, bool RESID>
__global__ __launch_bounds__(256) void gemm_wmma(
    const _Float16* __restrict__ A, int lda,
    const _Float16* __restrict__ Bt, int ldb,
    const float* __restrict__ bias,
    const float* __restrict__ resid,
    float* __restrict__ Cf, _Float16* __restrict__ Ch, int ldc, int K) {
  __shared__ __align__(16) _Float16 sA[2][128 * LDS_PITCH];
  __shared__ __align__(16) _Float16 sB[2][128 * LDS_PITCH];

  const int tid = threadIdx.x;
  const int lane = tid & 31, w = tid >> 5;
  const int wm = w >> 1, wn = w & 1;
  const int ml = lane & 15, hi = lane >> 4;
  const size_t m0 = (size_t)blockIdx.y * 128;
  const size_t n0 = (size_t)blockIdx.x * 128;

  f32x8 acc[2][4];
  for (int i = 0; i < 2; i++)
    for (int j = 0; j < 4; j++)
      for (int e = 0; e < 8; e++) acc[i][j][e] = 0.f;

  // staging map: thread -> (row = tid>>1, 16-half chunk = (tid&1)*16)
  const int sr = tid >> 1, sc = (tid & 1) * 16;
  const _Float16* gA = A + (m0 + sr) * (size_t)lda + sc;
  const _Float16* gB = Bt + (n0 + sr) * (size_t)ldb + sc;
  unsigned laoff[2], lboff[2];
  for (int b = 0; b < 2; b++) {
    laoff[b] = lds_off_of(&sA[b][sr * LDS_PITCH + sc]);
    lboff[b] = lds_off_of(&sB[b][sr * LDS_PITCH + sc]);
  }

  // prologue: stage K-slice 0 into buffer 0 (each thread: 2x16B for A, B)
  async_ld128(laoff[0],      gA);
  async_ld128(laoff[0] + 16, gA + 8);
  async_ld128(lboff[0],      gB);
  async_ld128(lboff[0] + 16, gB + 8);

  int buf = 0;
  for (int k0 = 0; k0 < K; k0 += 32) {
    wait_async0();        // our async loads into `buf` are complete
    __syncthreads();      // everyone's are; prior reads of buf^1 also retired
    if (k0 + 32 < K) {    // stream next K-slice into the other buffer
      const int nb = buf ^ 1;
      async_ld128(laoff[nb],      gA + k0 + 32);
      async_ld128(laoff[nb] + 16, gA + k0 + 40);
      async_ld128(lboff[nb],      gB + k0 + 32);
      async_ld128(lboff[nb] + 16, gB + k0 + 40);
    }

    f16x16 af[2], bf[4];
    for (int i = 0; i < 2; i++)
      af[i] = load_frag_a(&sA[buf][(wm * 32 + i * 16 + ml) * LDS_PITCH + hi * 8]);
    for (int j = 0; j < 4; j++)
      bf[j] = load_frag_b(&sB[buf][(wn * 64 + j * 16 + ml) * LDS_PITCH + hi * 8]);

    for (int i = 0; i < 2; i++)
      for (int j = 0; j < 4; j++)
        acc[i][j] = __builtin_amdgcn_wmma_f32_16x16x32_f16(
            false, af[i], false, bf[j], (short)0, acc[i][j], false, false);
    buf ^= 1;
  }

  // C layout: lanes 0-15 -> (M=e, N=lane); lanes 16-31 -> (M=8+e, N=lane-16)
  const int mh = hi * 8;
  for (int i = 0; i < 2; i++) {
    for (int j = 0; j < 4; j++) {
      const size_t gn = n0 + wn * 64 + j * 16 + ml;
      const float bb = bias[gn];
      for (int e = 0; e < 8; e++) {
        const size_t gm = m0 + wm * 32 + i * 16 + mh + e;
        float val = acc[i][j][e] + bb;
        if (ACT == 1) val = fmaxf(val, 0.f);
        if (ACT == 2) val = (val > 0.f) ? (val + 1.f) : __expf(val);  // elu(x)+1
        if (RESID) val += resid[gm * (size_t)ldc + gn];
        if (HALF_OUT) Ch[gm * (size_t)ldc + gn] = (_Float16)val;
        else          Cf[gm * (size_t)ldc + gn] = val;
      }
    }
  }
}

// ---------------------------------------------------------------------------
// Per-(b,h): kv[m,d] = sum_t v[t,m] * k[t,d]  (64x64, K=T=4096) + ksum[d]
// One block per (b,h). Tiles of k/v staged *transposed* into LDS so both
// WMMA operands read contiguous halves.
// ---------------------------------------------------------------------------
__global__ __launch_bounds__(256) void kv_ksum_kernel(
    const _Float16* __restrict__ k16, const _Float16* __restrict__ v16,
    _Float16* __restrict__ kv, float* __restrict__ ksum) {
  __shared__ __align__(16) _Float16 sVt[64 * LDS_PITCH];  // [m][t]
  __shared__ __align__(16) _Float16 sKt[64 * LDS_PITCH];  // [d][t]
  const int bh = blockIdx.x;
  const int b = bh >> 4, h = bh & 15;
  const _Float16* kb = k16 + (size_t)b * Tn * Dn + h * HDn;
  const _Float16* vb = v16 + (size_t)b * Tn * Dn + h * HDn;
  const int tid = threadIdx.x, lane = tid & 31, w = tid >> 5;
  const int ml = lane & 15, hi = lane >> 4;
  const int tm = w >> 1;          // wave row tile 0..3
  const int nb = (w & 1) * 2;     // wave covers col tiles nb, nb+1
  const int tt = tid >> 3;        // staging: t row 0..31
  const int mq = (tid & 7) * 8;   // staging: 8-half column chunk

  f32x8 acc[2];
  for (int j = 0; j < 2; j++)
    for (int e = 0; e < 8; e++) acc[j][e] = 0.f;
  float ks = 0.f;

  for (int t0 = 0; t0 < Tn; t0 += 32) {
    f16x8 vv = *(const f16x8*)(vb + (size_t)(t0 + tt) * Dn + mq);
    f16x8 kk = *(const f16x8*)(kb + (size_t)(t0 + tt) * Dn + mq);
    for (int e = 0; e < 8; e++) {
      sVt[(mq + e) * LDS_PITCH + tt] = vv[e];
      sKt[(mq + e) * LDS_PITCH + tt] = kk[e];
    }
    __syncthreads();
    if (tid < 64) {  // ksum[d] partial
      for (int t = 0; t < 32; t++) ks += (float)sKt[tid * LDS_PITCH + t];
    }
    f16x16 af = load_frag_a(sVt + (tm * 16 + ml) * LDS_PITCH + hi * 8);
    for (int j = 0; j < 2; j++) {
      f16x16 bf = load_frag_b(sKt + ((nb + j) * 16 + ml) * LDS_PITCH + hi * 8);
      acc[j] = __builtin_amdgcn_wmma_f32_16x16x32_f16(
          false, af, false, bf, (short)0, acc[j], false, false);
    }
    __syncthreads();
  }

  const int mh = hi * 8;
  for (int j = 0; j < 2; j++)
    for (int e = 0; e < 8; e++) {
      const int m = tm * 16 + mh + e;
      const int d = (nb + j) * 16 + ml;
      kv[(size_t)bh * HDn * HDn + m * HDn + d] = (_Float16)acc[j][e];
    }
  if (tid < 64) ksum[(size_t)bh * HDn + tid] = ks;
}

// ---------------------------------------------------------------------------
// z[bh*T + t] = 1 / (sum_d q[t,d] * ksum[d] + eps)
// ---------------------------------------------------------------------------
__global__ __launch_bounds__(256) void z_kernel(
    const _Float16* __restrict__ q16, const float* __restrict__ ksum,
    float* __restrict__ z) {
  const size_t id = (size_t)blockIdx.x * 256 + threadIdx.x;  // B*H*T total
  const int t = (int)(id % Tn);
  const int bh = (int)(id / Tn);
  const int b = bh >> 4, h = bh & 15;
  const _Float16* qr = q16 + ((size_t)(b * Tn + t)) * Dn + h * HDn;
  const float* kr = ksum + (size_t)bh * HDn;
  float s = 0.f;
  for (int d = 0; d < HDn; d++) s += (float)qr[d] * kr[d];
  z[id] = 1.f / (s + EPS_ATTN);
}

// ---------------------------------------------------------------------------
// Per-(b,h): out[t,m] = (sum_d q[t,d] * kv[m,d]) * z[t]   (M=T, N=64, K=64)
// kv row-major [m][d] is exactly the transposed-B layout -> direct global frags.
// ---------------------------------------------------------------------------
__global__ __launch_bounds__(256) void attn_kernel(
    const _Float16* __restrict__ q16, const _Float16* __restrict__ kv,
    const float* __restrict__ z, _Float16* __restrict__ attn) {
  const int bh = blockIdx.y, b = bh >> 4, h = bh & 15;
  const int m0 = blockIdx.x * 128;
  const int tid = threadIdx.x, lane = tid & 31, w = tid >> 5;
  const int ml = lane & 15, hi = lane >> 4;
  const _Float16* qb = q16 + (size_t)b * Tn * Dn + h * HDn;
  const _Float16* kvb = kv + (size_t)bh * HDn * HDn;

  f32x8 acc[4];
  for (int j = 0; j < 4; j++)
    for (int e = 0; e < 8; e++) acc[j][e] = 0.f;

  for (int k0 = 0; k0 < HDn; k0 += 32) {
    f16x16 af = load_frag_a(qb + (size_t)(m0 + w * 16 + ml) * Dn + k0 + hi * 8);
    for (int j = 0; j < 4; j++) {
      f16x16 bf = load_frag_b(kvb + (size_t)(j * 16 + ml) * HDn + k0 + hi * 8);
      acc[j] = __builtin_amdgcn_wmma_f32_16x16x32_f16(
          false, af, false, bf, (short)0, acc[j], false, false);
    }
  }

  const int mh = hi * 8;
  for (int j = 0; j < 4; j++)
    for (int e = 0; e < 8; e++) {
      const int row = m0 + w * 16 + mh + e;
      const float zz = z[(size_t)bh * Tn + row];
      attn[((size_t)(b * Tn + row)) * Dn + h * HDn + j * 16 + ml] =
          (_Float16)(acc[j][e] * zz);
    }
}

// ---------------------------------------------------------------------------
// LayerNorm over D=1024; one block (256 thr) per row; optional f32/f16 outputs
// ---------------------------------------------------------------------------
__global__ __launch_bounds__(256) void layernorm_kernel(
    const float* __restrict__ y, const float* __restrict__ g,
    const float* __restrict__ be, float* __restrict__ outf,
    _Float16* __restrict__ outh) {
  __shared__ float red[256];
  const int row = blockIdx.x, t = threadIdx.x;
  const float* p = y + (size_t)row * Dn;
  float v[4];
  float s = 0.f;
  for (int i = 0; i < 4; i++) { v[i] = p[t + i * 256]; s += v[i]; }
  red[t] = s; __syncthreads();
  for (int o = 128; o > 0; o >>= 1) { if (t < o) red[t] += red[t + o]; __syncthreads(); }
  const float mean = red[0] * (1.f / Dn);
  __syncthreads();
  float sv = 0.f;
  for (int i = 0; i < 4; i++) { const float d = v[i] - mean; sv += d * d; }
  red[t] = sv; __syncthreads();
  for (int o = 128; o > 0; o >>= 1) { if (t < o) red[t] += red[t + o]; __syncthreads(); }
  const float rstd = rsqrtf(red[0] * (1.f / Dn) + EPS_LN);
  for (int i = 0; i < 4; i++) {
    const int c = t + i * 256;
    const float o = (v[i] - mean) * rstd * g[c] + be[c];
    if (outf) outf[(size_t)row * Dn + c] = o;
    if (outh) outh[(size_t)row * Dn + c] = (_Float16)o;
  }
}

// ---------------------------------------------------------------------------
// Launch
// ---------------------------------------------------------------------------
extern "C" void kernel_launch(void* const* d_in, const int* in_sizes, int n_in,
                              void* d_out, int out_size, void* d_ws, size_t ws_size,
                              hipStream_t stream) {
  (void)in_sizes; (void)n_in; (void)out_size; (void)ws_size;
  const float* x   = (const float*)d_in[0];
  const float* Wq  = (const float*)d_in[1];
  const float* bq  = (const float*)d_in[2];
  const float* Wk  = (const float*)d_in[3];
  const float* bk  = (const float*)d_in[4];
  const float* Wv  = (const float*)d_in[5];
  const float* bv  = (const float*)d_in[6];
  const float* Wo  = (const float*)d_in[7];
  const float* bo  = (const float*)d_in[8];
  const float* g1  = (const float*)d_in[9];
  const float* be1 = (const float*)d_in[10];
  const float* W1  = (const float*)d_in[11];
  const float* b1  = (const float*)d_in[12];
  const float* W2  = (const float*)d_in[13];
  const float* b2  = (const float*)d_in[14];
  const float* g2  = (const float*)d_in[15];
  const float* be2 = (const float*)d_in[16];
  float* out = (float*)d_out;

  char* ws = (char*)d_ws;
  const size_t MB = 1ull << 20;
  // phase-reused workspace layout (peak ~346 MB)
  _Float16* WtQ  = (_Float16*)(ws + 0 * MB);    // 2MB each
  _Float16* WtK  = (_Float16*)(ws + 2 * MB);
  _Float16* WtV  = (_Float16*)(ws + 4 * MB);
  _Float16* WtO  = (_Float16*)(ws + 6 * MB);
  _Float16* W1t  = (_Float16*)(ws + 8 * MB);    // 8MB
  _Float16* W2t  = (_Float16*)(ws + 16 * MB);   // 8MB
  _Float16* x16  = (_Float16*)(ws + 24 * MB);   // 32MB (dead after QKV)
  _Float16* q16  = (_Float16*)(ws + 56 * MB);   // 32MB (dead after attn)
  _Float16* k16  = (_Float16*)(ws + 88 * MB);   // 32MB (dead after kv)
  _Float16* v16  = (_Float16*)(ws + 120 * MB);  // 32MB (dead after kv)
  float*    ksum = (float*)(ws + 152 * MB);                  // 16KB
  _Float16* kv16 = (_Float16*)(ws + 152 * MB + 64 * 1024);   // 512KB
  float*    zbuf = (float*)(ws + 153 * MB);                  // 1MB
  _Float16* at16 = (_Float16*)(ws + 154 * MB);  // 32MB (dead after Wo gemm)
  float*    y1   = (float*)(ws + 186 * MB);     // 64MB (dead after LN1)
  float*    x1f  = (float*)(ws + 250 * MB);     // 64MB
  _Float16* x1h  = (_Float16*)(ws + 314 * MB);  // 32MB
  _Float16* ff1  = (_Float16*)(ws + 24 * MB);   // 128MB, reuses x16/q16/k16/v16
  float*    y2   = (float*)(ws + 154 * MB);     // 64MB, reuses at16/y1

  const dim3 tb(32, 8);
  // 1. weight convert+transpose, x convert
  transpose_f32_to_f16<<<dim3(32, 32), tb, 0, stream>>>(Wq, WtQ, Dn, Dn);
  transpose_f32_to_f16<<<dim3(32, 32), tb, 0, stream>>>(Wk, WtK, Dn, Dn);
  transpose_f32_to_f16<<<dim3(32, 32), tb, 0, stream>>>(Wv, WtV, Dn, Dn);
  transpose_f32_to_f16<<<dim3(32, 32), tb, 0, stream>>>(Wo, WtO, Dn, Dn);
  transpose_f32_to_f16<<<dim3(128, 32), tb, 0, stream>>>(W1, W1t, Dn, FFn);
  transpose_f32_to_f16<<<dim3(32, 128), tb, 0, stream>>>(W2, W2t, FFn, Dn);
  f32_to_f16_kernel<<<(MTOK * Dn) / 1024, 256, 0, stream>>>(x, x16);

  // 2. QKV projections (q,k: elu+1; v: linear), all f16 outputs
  const dim3 gD(Dn / 128, MTOK / 128);
  gemm_wmma<2, true, false><<<gD, 256, 0, stream>>>(x16, Dn, WtQ, Dn, bq, nullptr, nullptr, q16, Dn, Dn);
  gemm_wmma<2, true, false><<<gD, 256, 0, stream>>>(x16, Dn, WtK, Dn, bk, nullptr, nullptr, k16, Dn, Dn);
  gemm_wmma<0, true, false><<<gD, 256, 0, stream>>>(x16, Dn, WtV, Dn, bv, nullptr, nullptr, v16, Dn, Dn);

  // 3. linear attention: kv + ksum, z, q@kv scaled by z
  kv_ksum_kernel<<<Bn * Hn, 256, 0, stream>>>(k16, v16, kv16, ksum);
  z_kernel<<<(Bn * Hn * Tn) / 256, 256, 0, stream>>>(q16, ksum, zbuf);
  attn_kernel<<<dim3(Tn / 128, Bn * Hn), 256, 0, stream>>>(q16, kv16, zbuf, at16);

  // 4. output projection + residual, LN1
  gemm_wmma<0, false, true><<<gD, 256, 0, stream>>>(at16, Dn, WtO, Dn, bo, x, y1, nullptr, Dn, Dn);
  layernorm_kernel<<<MTOK, 256, 0, stream>>>(y1, g1, be1, x1f, x1h);

  // 5. FFN + residual, LN2 -> d_out
  gemm_wmma<1, true, false><<<dim3(FFn / 128, MTOK / 128), 256, 0, stream>>>(
      x1h, Dn, W1t, Dn, b1, nullptr, nullptr, ff1, FFn, Dn);
  gemm_wmma<0, false, true><<<gD, 256, 0, stream>>>(ff1, FFn, W2t, FFn, b2, x1f, y2, nullptr, Dn, FFn);
  layernorm_kernel<<<MTOK, 256, 0, stream>>>(y2, g2, be2, out, nullptr);
}
